// PocketGVP_32701880992167
// MI455X (gfx1250) — matrix operand order repaired and checked by reading, not compiled, not run on hardware
//
#include <hip/hip_runtime.h>
#include <hip/hip_bf16.h>

typedef __attribute__((ext_vector_type(16))) _Float16 v16h;
typedef __attribute__((ext_vector_type(8)))  float    v8f;

// ---------------- problem constants ----------------
constexpr int   kN   = 32768;
constexpr int   kB   = 32;
constexpr int   kK   = 1024;
constexpr int   kE   = 524288;
constexpr int   kH   = 128;
constexpr float kEPS = 1e-8f;
constexpr float kLNE = 1e-5f;

// ---------------- workspace layout (bytes) ----------------
constexpr size_t SZ_S     = (size_t)kN * kH * 4;   // 16 MB  (node scalar feats)
constexpr size_t SZ_V     = (size_t)kN * 48 * 4;   //  6 MB  (node vector feats, 16x3)
constexpr size_t OFF_SA   = 0;
constexpr size_t OFF_SB   = OFF_SA + SZ_S;
constexpr size_t OFF_VA   = OFF_SB + SZ_S;
constexpr size_t OFF_VB   = OFF_VA + SZ_V;
constexpr size_t OFF_AGGS = OFF_VB + SZ_V;
constexpr size_t OFF_AGGV = OFF_AGGS + SZ_S;
constexpr size_t OFF_CNT  = OFF_AGGV + SZ_V;
constexpr size_t OFF_SCORE= OFF_CNT + (size_t)kN * 4;
constexpr size_t SZ_MSGW  = (size_t)6 * 8 * 512 * 2;   // 192x128 f16 frags
constexpr size_t SZ_UPDW  = (size_t)9 * 8 * 512 * 2;   // 288x128 f16 frags
constexpr size_t SZ_ATTNW = (size_t)4 * 8 * 512 * 2;   // 128x128 f16 frags
constexpr size_t OFF_MSGW = OFF_SCORE + (size_t)kN * 4;
constexpr size_t OFF_UPDW = OFF_MSGW + 3 * SZ_MSGW;
constexpr size_t OFF_ATTNW= OFF_UPDW + 3 * SZ_UPDW;

// ---------------- param leaf indices (jax tree flatten, dict keys sorted) ----------------
// 0 attn_b1, 1 attn_b2, 2 attn_w1, 3 attn_w2,
// conv l at 4+10l: +0 ln_b, +1 ln_g, +2 msg.Wh, +3 msg.Wmu, +4 msg.Ws_b, +5 msg.Ws_w,
//                  +6 upd.Wh, +7 upd.Wmu, +8 upd.Ws_b, +9 upd.Ws_w
// 34..37 in_gvp: Wh, Wmu, Ws_b, Ws_w; 38 node_s, 39 node_v, 40 edge_index, 41 edge_s,
// 42 edge_v, 43 batch

// Pack W (out 128 x Kdim, row-major) into WMMA B fragments: frag f = kc*8+nt holds the
// 32x16 tile (k = kc*32.., n = nt*16..) with lane = ((k/8)%2)*16 + n%16,
// element h -> k = (h/8)*16 + (lane/16)*8 + h%8 within chunk (matches 16-bit B layout in ISA 7.12.2).
__global__ void pack_b_frags(const float* __restrict__ W, _Float16* __restrict__ out,
                             int Kdim, int KPad) {
  int total = (KPad / 32) * 8 * 512;
  for (int g = blockIdx.x * blockDim.x + threadIdx.x; g < total; g += gridDim.x * blockDim.x) {
    int f = g >> 9; int r = g & 511; int lane = r >> 4; int h = r & 15;
    int kc = f >> 3, nt = f & 7;
    int n = nt * 16 + (lane & 15);
    int k = kc * 32 + (h >> 3) * 16 + (lane >> 4) * 8 + (h & 7);
    out[g] = (k < Kdim) ? (_Float16)W[n * Kdim + k] : (_Float16)0.f;
  }
}

__global__ void count_edges(const int* __restrict__ dst, float* __restrict__ cnt, int E) {
  int i = blockIdx.x * blockDim.x + threadIdx.x;
  if (i < E) atomicAdd(&cnt[dst[i]], 1.0f);
}

// Input GVP: (26,2x3) -> (128,16x3). Tiny matmul (42x128) -> plain VALU f32, 1 block/node.
__global__ __launch_bounds__(128) void input_gvp(
    const float* __restrict__ Wh, const float* __restrict__ Wmu,
    const float* __restrict__ Wsb, const float* __restrict__ Wsw,
    const float* __restrict__ node_s, const float* __restrict__ node_v,
    float* __restrict__ sOut, float* __restrict__ vOut) {
  __shared__ float shW[128 * 42];
  __shared__ float shIn[42];
  __shared__ float shVh[48];
  __shared__ float shVmu[48];
  __shared__ float shNmu[16];
  const int i = blockIdx.x, t = threadIdx.x;
  for (int j = t; j < 128 * 42; j += 128) shW[j] = Wsw[j];
  if (t < 26) shIn[t] = node_s[(size_t)i * 26 + t];
  if (t < 48) {
    int h = t / 3, c = t % 3;
    shVh[t] = Wh[h * 2 + 0] * node_v[(size_t)i * 6 + c] +
              Wh[h * 2 + 1] * node_v[(size_t)i * 6 + 3 + c];
  }
  __syncthreads();
  if (t < 16)
    shIn[26 + t] = sqrtf(shVh[t*3]*shVh[t*3] + shVh[t*3+1]*shVh[t*3+1] + shVh[t*3+2]*shVh[t*3+2]);
  if (t < 48) {
    int m = t / 3, c = t % 3;
    float a = 0.f;
#pragma unroll
    for (int h = 0; h < 16; ++h) a += Wmu[m * 16 + h] * shVh[h * 3 + c];
    shVmu[t] = a;
  }
  __syncthreads();
  float acc = Wsb[t];
#pragma unroll 6
  for (int j = 0; j < 42; ++j) acc += shIn[j] * shW[t * 42 + j];
  sOut[(size_t)i * 128 + t] = fmaxf(acc, 0.f);
  if (t < 16) {
    float n = sqrtf(shVmu[t*3]*shVmu[t*3] + shVmu[t*3+1]*shVmu[t*3+1] + shVmu[t*3+2]*shVmu[t*3+2]);
    shNmu[t] = fmaxf(n, kEPS);
  }
  __syncthreads();
  if (t < 48) {
    float n = shNmu[t / 3];
    vOut[(size_t)i * 48 + t] = shVmu[t] / n * (1.f / (1.f + expf(-n)));
  }
}

// Edge message GVP + scatter. 8 waves/block, 16 edges/wave, K=161 padded to 192.
__global__ __launch_bounds__(256) void edge_msg(
    const _Float16* __restrict__ Bfrag, const float* __restrict__ Wh,
    const float* __restrict__ Wmu, const float* __restrict__ Wsb,
    const int* __restrict__ src, const int* __restrict__ dst,
    const float* __restrict__ sIn, const float* __restrict__ vIn,
    const float* __restrict__ edge_s, const float* __restrict__ edge_v,
    float* __restrict__ aggS, float* __restrict__ aggV) {
  __shared__ __align__(32) _Float16 shB[6 * 8 * 512];   // 48 KB weight frags
  __shared__ __align__(32) _Float16 shA[8 * 16 * 192];  // 48 KB activation rows
  __shared__ float shWh[289];
  __shared__ float shWmu[272];
  __shared__ float shBias[128];
  __shared__ int shSrc[128], shDst[128];

  const int tid = threadIdx.x, lane = tid & 31, w = tid >> 5;
  {
    const unsigned* bs = (const unsigned*)Bfrag;
    unsigned* bd = (unsigned*)shB;
    for (int j = tid; j < 6 * 8 * 256; j += 256) bd[j] = bs[j];
  }
  for (int j = tid; j < 289; j += 256) shWh[j] = Wh[j];
  for (int j = tid; j < 272; j += 256) shWmu[j] = Wmu[j];
  if (tid < 128) shBias[tid] = Wsb[tid];
  const int eBase = blockIdx.x * 128;
  if (tid < 128) { shSrc[tid] = src[eBase + tid]; shDst[tid] = dst[eBase + tid]; }
  __syncthreads();

  const int aBase = w * 16 * 192;
  const int eW = eBase + w * 16;
  // coalesced gather of s[src] + edge_s -> f16 rows, zero-pad tail
  for (int i = 0; i < 64; ++i) {
    int idx = lane + 32 * i; int e = idx >> 7, c = idx & 127;
    shA[aBase + e * 192 + c] = (_Float16)sIn[(size_t)shSrc[w * 16 + e] * 128 + c];
  }
  for (int i = 0; i < 8; ++i) {
    int idx = lane + 32 * i; int e = idx >> 4, c = idx & 15;
    shA[aBase + e * 192 + 128 + c] = (_Float16)edge_s[(size_t)(eW + e) * 16 + c];
  }
  for (int i = 0; i < 16; ++i) {
    int idx = lane + 32 * i; int e = idx >> 5, c = idx & 31;
    shA[aBase + e * 192 + 160 + c] = (_Float16)0.f;
  }
  __syncthreads();
  // vector path (one lane per edge; fully unrolled so arrays stay in VGPRs)
  if (lane < 16) {
    const int e = lane, sN = shSrc[w * 16 + e], dN = shDst[w * 16 + e], eg = eW + e;
    float vc[51];
#pragma unroll
    for (int j = 0; j < 16; ++j)
#pragma unroll
      for (int c = 0; c < 3; ++c) vc[j * 3 + c] = vIn[(size_t)sN * 48 + j * 3 + c];
#pragma unroll
    for (int c = 0; c < 3; ++c) vc[48 + c] = edge_v[(size_t)eg * 3 + c];
    float mu[48];
#pragma unroll
    for (int q = 0; q < 48; ++q) mu[q] = 0.f;
#pragma unroll
    for (int h = 0; h < 17; ++h) {
      float v0 = 0.f, v1 = 0.f, v2 = 0.f;
#pragma unroll
      for (int j = 0; j < 17; ++j) {
        float wv = shWh[h * 17 + j];
        v0 += wv * vc[j * 3]; v1 += wv * vc[j * 3 + 1]; v2 += wv * vc[j * 3 + 2];
      }
      shA[aBase + e * 192 + 144 + h] = (_Float16)sqrtf(v0 * v0 + v1 * v1 + v2 * v2);
#pragma unroll
      for (int m = 0; m < 16; ++m) {
        float wm = shWmu[m * 17 + h];
        mu[m * 3] += wm * v0; mu[m * 3 + 1] += wm * v1; mu[m * 3 + 2] += wm * v2;
      }
    }
#pragma unroll
    for (int m = 0; m < 16; ++m) {
      float n = sqrtf(mu[m*3]*mu[m*3] + mu[m*3+1]*mu[m*3+1] + mu[m*3+2]*mu[m*3+2]);
      n = fmaxf(n, kEPS);
      float sc = (1.f / (1.f + expf(-n))) / n;
      atomicAdd(&aggV[(size_t)dN * 48 + m * 3 + 0], mu[m * 3 + 0] * sc);
      atomicAdd(&aggV[(size_t)dN * 48 + m * 3 + 1], mu[m * 3 + 1] * sc);
      atomicAdd(&aggV[(size_t)dN * 48 + m * 3 + 2], mu[m * 3 + 2] * sc);
    }
  }
  __syncthreads();
  // WMMA: (16 edges x 192) @ (192 x 128)
  v8f acc[8] = {};
  const int row = lane & 15, hi = lane >> 4;
#pragma unroll
  for (int kc = 0; kc < 6; ++kc) {
    union { v16h v; unsigned u[8]; } a;
#pragma unroll
    for (int hp = 0; hp < 8; ++hp) {
      int k0 = kc * 32 + ((hp >> 2) << 4) + (hi << 3) + ((hp & 3) << 1);
      a.u[hp] = *(const unsigned*)&shA[aBase + row * 192 + k0];
    }
    const _Float16* bp = &shB[(kc * 8) * 512 + lane * 16];
#pragma unroll
    for (int nt = 0; nt < 8; ++nt) {
      v16h b = *(const v16h*)(bp + nt * 512);
      acc[nt] = __builtin_amdgcn_wmma_f32_16x16x32_f16(false, a.v, false, b,
                                                       (short)0, acc[nt], false, false);
    }
  }
  // bias + relu + scatter-add (L2-resident aggregation buffers)
#pragma unroll
  for (int nt = 0; nt < 8; ++nt) {
    const int ch = nt * 16 + row;
    const float bsv = shBias[ch];
#pragma unroll
    for (int r = 0; r < 8; ++r) {
      const int m = r + (hi << 3);
      const int dN = shDst[w * 16 + m];
      atomicAdd(&aggS[(size_t)dN * 128 + ch], fmaxf(acc[nt][r] + bsv, 0.f));
    }
  }
}

// Node update GVP + residual + LayerNorm. 4 waves/block, 16 nodes/wave, K=288.
__global__ __launch_bounds__(128) void node_update(
    const _Float16* __restrict__ Bfrag, const float* __restrict__ Wh,
    const float* __restrict__ Wmu, const float* __restrict__ Wsb,
    const float* __restrict__ lnG, const float* __restrict__ lnB,
    const float* __restrict__ sIn, const float* __restrict__ vIn,
    const float* __restrict__ aggS, const float* __restrict__ aggV,
    const float* __restrict__ cnt,
    float* __restrict__ sOut, float* __restrict__ vOut) {
  __shared__ __align__(32) _Float16 shB[9 * 8 * 512];   // 72 KB
  __shared__ __align__(32) _Float16 shA[4 * 16 * 288];  // 36 KB
  __shared__ float shVcat[64 * 96];                     // 24 KB
  __shared__ float shWh[1024];
  __shared__ float shWmu[512];
  __shared__ float shBias[128], shG[128], shBeta[128];

  const int tid = threadIdx.x, lane = tid & 31, w = tid >> 5;
  {
    const unsigned* bs = (const unsigned*)Bfrag;
    unsigned* bd = (unsigned*)shB;
    for (int j = tid; j < 9 * 8 * 256; j += 128) bd[j] = bs[j];
  }
  for (int j = tid; j < 1024; j += 128) shWh[j] = Wh[j];
  for (int j = tid; j < 512; j += 128) shWmu[j] = Wmu[j];
  if (tid < 128) { shBias[tid] = Wsb[tid]; shG[tid] = lnG[tid]; shBeta[tid] = lnB[tid]; }
  const int nW = blockIdx.x * 64 + w * 16;
  const int aBase = w * 16 * 288;
  for (int i = 0; i < 128; ++i) {
    int idx = lane + 32 * i; int e = idx >> 8, c = idx & 255;
    int node = nW + e;
    float ivc = 1.f / fmaxf(cnt[node], 1.f);
    float val = (c < 128) ? sIn[(size_t)node * 128 + c]
                          : aggS[(size_t)node * 128 + (c - 128)] * ivc;
    shA[aBase + e * 288 + c] = (_Float16)val;
  }
  for (int i = 0; i < 48; ++i) {
    int idx = lane + 32 * i; int e = idx / 96, j = idx - 96 * e;
    int node = nW + e;
    float ivc = 1.f / fmaxf(cnt[node], 1.f);
    float val = (j < 48) ? vIn[(size_t)node * 48 + j]
                         : aggV[(size_t)node * 48 + (j - 48)] * ivc;
    shVcat[(w * 16 + e) * 96 + j] = val;
  }
  __syncthreads();
  // vector path: lane pair per node, Vh rows split 16/16, Vmu combined via shfl_xor
  {
    const int e = lane >> 1, sub = lane & 1;
    const int node = nW + e;
    const float* vc = &shVcat[(w * 16 + e) * 96];
    float mu[48];
#pragma unroll
    for (int q = 0; q < 48; ++q) mu[q] = 0.f;
#pragma unroll
    for (int hh = 0; hh < 16; ++hh) {
      const int h = sub * 16 + hh;
      float v0 = 0.f, v1 = 0.f, v2 = 0.f;
      for (int j = 0; j < 32; ++j) {
        float wv = shWh[h * 32 + j];
        v0 += wv * vc[j * 3]; v1 += wv * vc[j * 3 + 1]; v2 += wv * vc[j * 3 + 2];
      }
      shA[aBase + e * 288 + 256 + h] = (_Float16)sqrtf(v0 * v0 + v1 * v1 + v2 * v2);
#pragma unroll
      for (int m = 0; m < 16; ++m) {
        float wm = shWmu[m * 32 + h];
        mu[m * 3] += wm * v0; mu[m * 3 + 1] += wm * v1; mu[m * 3 + 2] += wm * v2;
      }
    }
#pragma unroll
    for (int q = 0; q < 48; ++q) mu[q] += __shfl_xor(mu[q], 1, 32);
    const int m0 = sub * 8;
#pragma unroll
    for (int mm = 0; mm < 8; ++mm) {
      const int m = m0 + mm;
      float n = sqrtf(mu[m*3]*mu[m*3] + mu[m*3+1]*mu[m*3+1] + mu[m*3+2]*mu[m*3+2]);
      n = fmaxf(n, kEPS);
      float sc = (1.f / (1.f + expf(-n))) / n;
#pragma unroll
      for (int c = 0; c < 3; ++c)
        vOut[(size_t)node * 48 + m * 3 + c] =
            vIn[(size_t)node * 48 + m * 3 + c] + mu[m * 3 + c] * sc;
    }
  }
  __syncthreads();
  // WMMA: (16 nodes x 288) @ (288 x 128)
  v8f acc[8] = {};
  const int row = lane & 15, hi = lane >> 4;
#pragma unroll
  for (int kc = 0; kc < 9; ++kc) {
    union { v16h v; unsigned u[8]; } a;
#pragma unroll
    for (int hp = 0; hp < 8; ++hp) {
      int k0 = kc * 32 + ((hp >> 2) << 4) + (hi << 3) + ((hp & 3) << 1);
      a.u[hp] = *(const unsigned*)&shA[aBase + row * 288 + k0];
    }
    const _Float16* bp = &shB[(kc * 8) * 512 + lane * 16];
#pragma unroll
    for (int nt = 0; nt < 8; ++nt) {
      v16h b = *(const v16h*)(bp + nt * 512);
      acc[nt] = __builtin_amdgcn_wmma_f32_16x16x32_f16(false, a.v, false, b,
                                                       (short)0, acc[nt], false, false);
    }
  }
  // bias + relu + residual, then LayerNorm via cross-lane reductions on C fragments
#pragma unroll
  for (int nt = 0; nt < 8; ++nt) {
    const int ch = nt * 16 + row;
#pragma unroll
    for (int r = 0; r < 8; ++r) {
      const int node = nW + r + (hi << 3);
      acc[nt][r] = sIn[(size_t)node * 128 + ch] + fmaxf(acc[nt][r] + shBias[ch], 0.f);
    }
  }
#pragma unroll
  for (int r = 0; r < 8; ++r) {
    float sm = 0.f;
#pragma unroll
    for (int nt = 0; nt < 8; ++nt) sm += acc[nt][r];
    sm += __shfl_xor(sm, 1, 32); sm += __shfl_xor(sm, 2, 32);
    sm += __shfl_xor(sm, 4, 32); sm += __shfl_xor(sm, 8, 32);
    const float mean = sm * (1.f / 128.f);
    float vs = 0.f;
#pragma unroll
    for (int nt = 0; nt < 8; ++nt) { float d = acc[nt][r] - mean; vs += d * d; }
    vs += __shfl_xor(vs, 1, 32); vs += __shfl_xor(vs, 2, 32);
    vs += __shfl_xor(vs, 4, 32); vs += __shfl_xor(vs, 8, 32);
    const float rs = rsqrtf(vs * (1.f / 128.f) + kLNE);
    const int node = nW + r + (hi << 3);
#pragma unroll
    for (int nt = 0; nt < 8; ++nt) {
      const int ch = nt * 16 + row;
      sOut[(size_t)node * 128 + ch] = (acc[nt][r] - mean) * rs * shG[ch] + shBeta[ch];
    }
  }
}

// Attention scores: tanh(s @ w1^T + b1) . w2 + b2, via WMMA + shuffle dot.
__global__ __launch_bounds__(256) void attn_scores(
    const _Float16* __restrict__ Bfrag, const float* __restrict__ b1,
    const float* __restrict__ w2, const float* __restrict__ b2,
    const float* __restrict__ sIn, float* __restrict__ scores) {
  __shared__ __align__(32) _Float16 shB[4 * 8 * 512];
  __shared__ __align__(32) _Float16 shA[8 * 16 * 128];
  __shared__ float shBias[128], shW2[128];
  const int tid = threadIdx.x, lane = tid & 31, w = tid >> 5;
  {
    const unsigned* bs = (const unsigned*)Bfrag;
    unsigned* bd = (unsigned*)shB;
    for (int j = tid; j < 4 * 8 * 256; j += 256) bd[j] = bs[j];
  }
  if (tid < 128) { shBias[tid] = b1[tid]; shW2[tid] = w2[tid]; }
  const int nW = blockIdx.x * 128 + w * 16;
  const int aBase = w * 16 * 128;
  for (int i = 0; i < 64; ++i) {
    int idx = lane + 32 * i; int e = idx >> 7, c = idx & 127;
    shA[aBase + e * 128 + c] = (_Float16)sIn[(size_t)(nW + e) * 128 + c];
  }
  __syncthreads();
  v8f acc[8] = {};
  const int row = lane & 15, hi = lane >> 4;
#pragma unroll
  for (int kc = 0; kc < 4; ++kc) {
    union { v16h v; unsigned u[8]; } a;
#pragma unroll
    for (int hp = 0; hp < 8; ++hp) {
      int k0 = kc * 32 + ((hp >> 2) << 4) + (hi << 3) + ((hp & 3) << 1);
      a.u[hp] = *(const unsigned*)&shA[aBase + row * 128 + k0];
    }
    const _Float16* bp = &shB[(kc * 8) * 512 + lane * 16];
#pragma unroll
    for (int nt = 0; nt < 8; ++nt) {
      v16h b = *(const v16h*)(bp + nt * 512);
      acc[nt] = __builtin_amdgcn_wmma_f32_16x16x32_f16(false, a.v, false, b,
                                                       (short)0, acc[nt], false, false);
    }
  }
  const float b2v = b2[0];
#pragma unroll
  for (int r = 0; r < 8; ++r) {
    float p = 0.f;
#pragma unroll
    for (int nt = 0; nt < 8; ++nt) {
      const int ch = nt * 16 + row;
      p += tanhf(acc[nt][r] + shBias[ch]) * shW2[ch];
    }
    p += __shfl_xor(p, 1, 32); p += __shfl_xor(p, 2, 32);
    p += __shfl_xor(p, 4, 32); p += __shfl_xor(p, 8, 32);
    if (row == 0) scores[nW + r + (hi << 3)] = p + b2v;
  }
}

// tokens (dense batch, bijective since N = B*K and batch is sorted uniform) + mask.
__global__ void write_tokens_mask(const float* __restrict__ sFin, const int* __restrict__ batch,
                                  float* __restrict__ tokens, float* __restrict__ maskOut) {
  const int total = kN * kH + kN;
  for (int g = blockIdx.x * blockDim.x + threadIdx.x; g < total; g += gridDim.x * blockDim.x) {
    if (g < kN * kH) {
      int i = g >> 7, c = g & 127;
      int b = batch[i]; int pos = i - b * kK;
      tokens[((size_t)b * kK + pos) * kH + c] = sFin[g];
    } else {
      int i = g - kN * kH;
      int b = batch[i]; int pos = i - b * kK;
      maskOut[(size_t)b * kK + pos] = 1.0f;
    }
  }
}

// Per-graph softmax + weighted pool.
__global__ __launch_bounds__(256) void attn_pool(
    const float* __restrict__ scores, const float* __restrict__ sFin,
    float* __restrict__ pool) {
  __shared__ float red[256];
  const int b = blockIdx.x, tid = threadIdx.x;
  const int base = b * kK;
  float m = -3.402823466e38f;
  for (int i = tid; i < kK; i += 256) m = fmaxf(m, scores[base + i]);
  red[tid] = m; __syncthreads();
  for (int s = 128; s > 0; s >>= 1) {
    if (tid < s) red[tid] = fmaxf(red[tid], red[tid + s]);
    __syncthreads();
  }
  const float mx = red[0]; __syncthreads();
  float sum = 0.f;
  for (int i = tid; i < kK; i += 256) sum += expf(scores[base + i] - mx);
  red[tid] = sum; __syncthreads();
  for (int s = 128; s > 0; s >>= 1) {
    if (tid < s) red[tid] += red[tid + s];
    __syncthreads();
  }
  const float invZ = 1.f / red[0]; __syncthreads();
  const int g = tid >> 7, ch = tid & 127;
  float acc = 0.f;
  for (int i = g; i < kK; i += 2)
    acc += expf(scores[base + i] - mx) * invZ * sFin[(size_t)(base + i) * kH + ch];
  red[tid] = acc; __syncthreads();
  if (g == 0) pool[(size_t)b * kH + ch] = red[ch] + red[128 + ch];
}

extern "C" void kernel_launch(void* const* d_in, const int* in_sizes, int n_in,
                              void* d_out, int out_size, void* d_ws, size_t ws_size,
                              hipStream_t stream) {
  auto F = [&](int i) { return (const float*)d_in[i]; };
  char* ws = (char*)d_ws;
  float* sA    = (float*)(ws + OFF_SA);
  float* sBuf  = (float*)(ws + OFF_SB);
  float* vA    = (float*)(ws + OFF_VA);
  float* vBuf  = (float*)(ws + OFF_VB);
  float* aggS  = (float*)(ws + OFF_AGGS);
  float* aggV  = (float*)(ws + OFF_AGGV);
  float* cnt   = (float*)(ws + OFF_CNT);
  float* score = (float*)(ws + OFF_SCORE);
  _Float16* msgW[3]; _Float16* updW[3];
  for (int l = 0; l < 3; ++l) {
    msgW[l] = (_Float16*)(ws + OFF_MSGW + (size_t)l * SZ_MSGW);
    updW[l] = (_Float16*)(ws + OFF_UPDW + (size_t)l * SZ_UPDW);
  }
  _Float16* attnW = (_Float16*)(ws + OFF_ATTNW);

  const int* eidx = (const int*)d_in[40];
  const int* src = eidx;
  const int* dst = eidx + kE;
  const int* batch = (const int*)d_in[43];
  const float* node_s = F(38); const float* node_v = F(39);
  const float* edge_s = F(41); const float* edge_v = F(42);

  // weight packing (f32 -> f16 WMMA B-fragments)
  for (int l = 0; l < 3; ++l) {
    const int cb = 4 + l * 10;
    pack_b_frags<<<96, 256, 0, stream>>>(F(cb + 5), msgW[l], 161, 192);
    pack_b_frags<<<144, 256, 0, stream>>>(F(cb + 9), updW[l], 288, 288);
  }
  pack_b_frags<<<64, 256, 0, stream>>>(F(2), attnW, 128, 128);

  // in-degree counts (identical across layers)
  hipMemsetAsync(cnt, 0, (size_t)kN * 4, stream);
  count_edges<<<kE / 256, 256, 0, stream>>>(dst, cnt, kE);

  input_gvp<<<kN, 128, 0, stream>>>(F(34), F(35), F(36), F(37), node_s, node_v, sA, vA);

  float* sCur = sA;   float* vCur = vA;
  float* sNxt = sBuf; float* vNxt = vBuf;
  for (int l = 0; l < 3; ++l) {
    const int cb = 4 + l * 10;
    hipMemsetAsync(aggS, 0, SZ_S, stream);
    hipMemsetAsync(aggV, 0, SZ_V, stream);
    edge_msg<<<kE / 128, 256, 0, stream>>>(msgW[l], F(cb + 2), F(cb + 3), F(cb + 4),
                                           src, dst, sCur, vCur, edge_s, edge_v, aggS, aggV);
    node_update<<<kN / 64, 128, 0, stream>>>(updW[l], F(cb + 6), F(cb + 7), F(cb + 8),
                                             F(cb + 1), F(cb + 0),
                                             sCur, vCur, aggS, aggV, cnt, sNxt, vNxt);
    float* t;
    t = sCur; sCur = sNxt; sNxt = t;
    t = vCur; vCur = vNxt; vNxt = t;
  }

  attn_scores<<<kN / 128, 256, 0, stream>>>(attnW, F(0), F(3), F(1), sCur, score);

  float* tokens  = (float*)d_out;
  float* pool    = tokens + (size_t)kN * kH;
  float* maskOut = pool + (size_t)kB * kH;
  write_tokens_mask<<<2048, 256, 0, stream>>>(sCur, batch, tokens, maskOut);
  attn_pool<<<kB, 256, 0, stream>>>(score, sCur, pool);

  (void)in_sizes; (void)n_in; (void)out_size; (void)ws_size;
}
